// CrossAttention_61538291417908
// MI455X (gfx1250) — matrix operand order repaired and checked by reading, not compiled, not run on hardware
//
#include <hip/hip_runtime.h>

#define DIMC 1024
#define HEADS 16
#define HEAD_DIM 64
#define BATCH 4
#define SEQ_N 2048
#define SEQ_M 2048

typedef __attribute__((ext_vector_type(16))) __bf16 v16bf;
typedef __attribute__((ext_vector_type(8)))  float  v8f;
typedef __attribute__((ext_vector_type(4)))  unsigned int u32x4;

__device__ __forceinline__ unsigned short f2bf(float f) {
    unsigned int u = __float_as_uint(f);
    u += 0x7FFFu + ((u >> 16) & 1u);      // round-to-nearest-even
    return (unsigned short)(u >> 16);
}

union FragU { v16bf v; u32x4 q[2]; };

// A fragment, 16x32 bf16 (M x K), row-major source, row stride `ld` elements.
// ISA layout: lanes 0-15 hold row M=lane, K = {0..7, 16..23}; lanes 16-31 K = {8..15, 24..31}.
__device__ __forceinline__ v16bf load_a_frag(const unsigned short* base, int ld, int lane) {
    const int m    = lane & 15;
    const int ksel = (lane & 16) ? 8 : 0;
    FragU u;
    const unsigned short* p = base + m * ld + ksel;
    u.q[0] = *(const u32x4*)(p);        // K = ksel .. ksel+7
    u.q[1] = *(const u32x4*)(p + 16);   // K = ksel+16 .. ksel+23
    return u.v;
}

// B fragment, 32x16 bf16 (K x N). Source: column n contiguous in K with stride `ld`
// between columns. ISA layout: lanes 0-15 hold col N=lane, K = {0..7, 16..23}; lanes 16-31 K = {8..15, 24..31}.
__device__ __forceinline__ v16bf load_b_frag(const unsigned short* base, int ld, int lane) {
    const int n    = lane & 15;
    const int koff = (lane & 16) ? 8 : 0;
    FragU u;
    const unsigned short* p = base + (long)n * ld + koff;
    u.q[0] = *(const u32x4*)(p);        // K = koff .. koff+7
    u.q[1] = *(const u32x4*)(p + 16);   // K = koff+16 .. koff+23
    return u.v;
}

__device__ __forceinline__ v8f wmma_bf16(v16bf a, v16bf b, v8f c) {
    return __builtin_amdgcn_wmma_f32_16x16x32_bf16(
        /*neg_a=*/false, a, /*neg_b=*/false, b,
        /*c_mod=*/(short)0, c, /*reuse_a=*/false, /*reuse_b=*/false);
}

// ---------------------------------------------------------------- cast kernel
__global__ __launch_bounds__(256) void cast_f32_bf16(const float* __restrict__ in,
                                                     unsigned short* __restrict__ out,
                                                     int n) {
    int i = blockIdx.x * blockDim.x + threadIdx.x;
    if (i < n) out[i] = f2bf(in[i]);
}

// ------------------------------------------------------------- projection GEMM
// out = (A[T,1024] @ W[1024,1024]^T + bias) * oscale.
// 8 waves/block; block tile 128 rows x 64 cols; W tile staged in LDS per k-step
// (loaded once per block, read as B-fragments by all 8 waves via ds_load_b128).
// MODE 0: bf16 out, head layout [B,H,L,64]
// MODE 1: bf16 out, transposed head layout [B,H,64,L]   (for V)
// MODE 2: fp32 out, plain [T,1024]                      (final projection)
#define WLDS_LD 40   // 64 cols x 40 (32 data + 8 pad) -> 16 distinct banks per frag read
template <int MODE>
__global__ __launch_bounds__(256) void proj_gemm(const unsigned short* __restrict__ A,
                                                 const unsigned short* __restrict__ W,
                                                 const float* __restrict__ bias,
                                                 void* __restrict__ out,
                                                 float oscale) {
    __shared__ unsigned short Wlds[64 * WLDS_LD];   // 5 KB

    const int tid  = threadIdx.x;
    const int lane = tid & 31;
    const int wave = tid >> 5;
    const int row0 = blockIdx.x * 128 + wave * 16;
    const int col0 = blockIdx.y * 64;

    // cooperative W-tile copy mapping: each thread owns 8 contiguous elements of one column
    const int ccol = tid >> 2;           // 0..63
    const int ck   = (tid & 3) * 8;      // 0,8,16,24
    const unsigned short* wsrc = W + (long)(col0 + ccol) * DIMC + ck;

    v8f acc[4] = {};
    const unsigned short* Arow = A + (long)row0 * DIMC;

    for (int kb = 0; kb < DIMC; kb += 32) {
        u32x4 wv = *(const u32x4*)(wsrc + kb);          // global, 16B per thread
        __syncthreads();                                // previous-iter reads done
        *(u32x4*)(&Wlds[ccol * WLDS_LD + ck]) = wv;     // stage tile
        __syncthreads();

        v16bf a  = load_a_frag(Arow + kb, DIMC, lane);  // per-wave, not redundant
        v16bf b0 = load_b_frag(&Wlds[ 0 * WLDS_LD], WLDS_LD, lane);
        v16bf b1 = load_b_frag(&Wlds[16 * WLDS_LD], WLDS_LD, lane);
        v16bf b2 = load_b_frag(&Wlds[32 * WLDS_LD], WLDS_LD, lane);
        v16bf b3 = load_b_frag(&Wlds[48 * WLDS_LD], WLDS_LD, lane);
        acc[0] = wmma_bf16(a, b0, acc[0]);
        acc[1] = wmma_bf16(a, b1, acc[1]);
        acc[2] = wmma_bf16(a, b2, acc[2]);
        acc[3] = wmma_bf16(a, b3, acc[3]);
    }

    const int nl   = lane & 15;
    const int mofs = (lane & 16) ? 8 : 0;
#pragma unroll
    for (int t = 0; t < 4; ++t) {
        const int n_global = col0 + t * 16 + nl;
        const float bv_ = bias[n_global];
#pragma unroll
        for (int r = 0; r < 8; ++r) {
            const int   tg  = row0 + r + mofs;          // token row
            const float val = (acc[t][r] + bv_) * oscale;
            if (MODE == 2) {
                ((float*)out)[(long)tg * DIMC + n_global] = val;
            } else {
                const int b_ = tg >> 11;                // / 2048
                const int l_ = tg & 2047;
                const int h  = n_global >> 6;
                const int d  = n_global & 63;
                unsigned short* o = (unsigned short*)out;
                if (MODE == 0)
                    o[((long)(b_ * HEADS + h) * SEQ_M + l_) * HEAD_DIM + d] = f2bf(val);
                else
                    o[((long)(b_ * HEADS + h) * HEAD_DIM + d) * SEQ_M + l_] = f2bf(val);
            }
        }
    }
}

// ------------------------------------------------------------- flash attention
// Q [B,H,N,64] bf16 (pre-scaled by D^-0.5), K [B,H,M,64] bf16, VT [B,H,64,M] bf16
// -> AO [B,N,1024] bf16.  Each wave owns a 16-row Q tile and streams M in
// 64-column chunks with online softmax (reductions amortized over 4 S-tiles).
#define PLDS_LD 72   // 64 data cols + 8 pad -> conflict-free 16-lane column reads
__global__ __launch_bounds__(256) void flash_attn(const unsigned short* __restrict__ Q,
                                                  const unsigned short* __restrict__ K,
                                                  const unsigned short* __restrict__ VT,
                                                  unsigned short* __restrict__ AO) {
    __shared__ unsigned short ldsP[8][16 * PLDS_LD];   // per-wave P tile (16 x 64 bf16), 18 KB

    const int lane = threadIdx.x & 31;
    const int wave = threadIdx.x >> 5;
    const int bh   = blockIdx.y;
    const int b    = bh >> 4;
    const int h    = bh & 15;
    const int row0 = blockIdx.x * 128 + wave * 16;

    const unsigned short* Qb = Q  + ((long)bh * SEQ_N + row0) * HEAD_DIM;
    const unsigned short* Kb = K  + (long)bh * SEQ_M * HEAD_DIM;
    const unsigned short* Vb = VT + (long)bh * HEAD_DIM * SEQ_M;
    unsigned short* ldsW = &ldsP[wave][0];

    // resident Q fragments (K = 0..31 and 32..63); Q already carries the 1/8 scale
    const v16bf qa0 = load_a_frag(Qb + 0,  HEAD_DIM, lane);
    const v16bf qa1 = load_a_frag(Qb + 32, HEAD_DIM, lane);

    v8f   o[4] = {};
    float m_r[8], l_r[8];
#pragma unroll
    for (int r = 0; r < 8; ++r) { m_r[r] = -1e30f; l_r[r] = 0.0f; }

    for (int c = 0; c < SEQ_M; c += 64) {
        __builtin_prefetch(Kb + (long)(c + 64) * HEAD_DIM, 0, 1);   // global_prefetch_b8

        // S = Q K^T : four 16x16 column tiles, fp32 accum
        v8f s[4] = {};
#pragma unroll
        for (int t = 0; t < 4; ++t) {
            const unsigned short* kp = Kb + (long)(c + 16 * t) * HEAD_DIM;
            s[t] = wmma_bf16(qa0, load_b_frag(kp + 0,  HEAD_DIM, lane), s[t]);
            s[t] = wmma_bf16(qa1, load_b_frag(kp + 32, HEAD_DIM, lane), s[t]);
        }

#pragma unroll
        for (int r = 0; r < 8; ++r) {
            float x0 = s[0][r], x1 = s[1][r], x2 = s[2][r], x3 = s[3][r];
            // row max across the 16-lane half (row r for lanes 0-15, row r+8 for 16-31)
            float mx = fmaxf(fmaxf(x0, x1), fmaxf(x2, x3));
            mx = fmaxf(mx, __shfl_xor(mx, 1, 32));
            mx = fmaxf(mx, __shfl_xor(mx, 2, 32));
            mx = fmaxf(mx, __shfl_xor(mx, 4, 32));
            mx = fmaxf(mx, __shfl_xor(mx, 8, 32));
            float mnew  = fmaxf(m_r[r], mx);
            float alpha = __expf(m_r[r] - mnew);
            float p0 = __expf(x0 - mnew);
            float p1 = __expf(x1 - mnew);
            float p2 = __expf(x2 - mnew);
            float p3 = __expf(x3 - mnew);
            float rs = (p0 + p1) + (p2 + p3);
            rs += __shfl_xor(rs, 1, 32);
            rs += __shfl_xor(rs, 2, 32);
            rs += __shfl_xor(rs, 4, 32);
            rs += __shfl_xor(rs, 8, 32);
            l_r[r] = l_r[r] * alpha + rs;
            m_r[r] = mnew;
#pragma unroll
            for (int t = 0; t < 4; ++t) o[t][r] *= alpha;
            // transpose P to A-fragment layout via per-wave LDS
            const int prow = r + ((lane & 16) ? 8 : 0);
            const int cl   = lane & 15;
            ldsW[prow * PLDS_LD + cl]      = f2bf(p0);
            ldsW[prow * PLDS_LD + 16 + cl] = f2bf(p1);
            ldsW[prow * PLDS_LD + 32 + cl] = f2bf(p2);
            ldsW[prow * PLDS_LD + 48 + cl] = f2bf(p3);
        }
        // per-wave LDS is in-order; wait for stores, block compiler reordering
        asm volatile("s_wait_dscnt 0x0" ::: "memory");
        v16bf pa0 = load_a_frag(ldsW + 0,  PLDS_LD, lane);   // P cols 0..31
        v16bf pa1 = load_a_frag(ldsW + 32, PLDS_LD, lane);   // P cols 32..63
#pragma unroll
        for (int t = 0; t < 4; ++t) {
            const unsigned short* vp = Vb + (long)(t * 16) * SEQ_M + c;
            o[t] = wmma_bf16(pa0, load_b_frag(vp + 0,  SEQ_M, lane), o[t]);
            o[t] = wmma_bf16(pa1, load_b_frag(vp + 32, SEQ_M, lane), o[t]);
        }
        asm volatile("s_wait_dscnt 0x0" ::: "memory");
    }

    const int nl   = lane & 15;
    const int mofs = (lane & 16) ? 8 : 0;
#pragma unroll
    for (int t = 0; t < 4; ++t) {
#pragma unroll
        for (int r = 0; r < 8; ++r) {
            const float val = o[t][r] / l_r[r];
            const int tok = row0 + r + mofs;
            const int d   = t * 16 + nl;
            AO[(long)(b * SEQ_N + tok) * DIMC + h * HEAD_DIM + d] = f2bf(val);
        }
    }
}

// --------------------------------------------------------------------- launch
extern "C" void kernel_launch(void* const* d_in, const int* in_sizes, int n_in,
                              void* d_out, int out_size, void* d_ws, size_t ws_size,
                              hipStream_t stream) {
    (void)in_sizes; (void)n_in; (void)out_size; (void)ws_size;
    const float* x   = (const float*)d_in[0];
    const float* ctx = (const float*)d_in[1];
    const float* Wq  = (const float*)d_in[2];
    const float* bq  = (const float*)d_in[3];
    const float* Wk  = (const float*)d_in[4];
    const float* bk  = (const float*)d_in[5];
    const float* Wv  = (const float*)d_in[6];
    const float* bv  = (const float*)d_in[7];
    const float* Wo  = (const float*)d_in[8];
    const float* bo  = (const float*)d_in[9];
    float* out = (float*)d_out;

    char* ws = (char*)d_ws;
    const size_t MB = 1024u * 1024u;
    unsigned short* xb  = (unsigned short*)(ws + 0 * MB);   // [8192,1024] bf16
    unsigned short* cb  = (unsigned short*)(ws + 16 * MB);
    unsigned short* wqb = (unsigned short*)(ws + 32 * MB);  // [1024,1024] bf16
    unsigned short* wkb = (unsigned short*)(ws + 34 * MB);
    unsigned short* wvb = (unsigned short*)(ws + 36 * MB);
    unsigned short* wob = (unsigned short*)(ws + 38 * MB);
    unsigned short* Qh  = (unsigned short*)(ws + 40 * MB);  // [B,H,N,64]  (pre-scaled)
    unsigned short* Kh  = (unsigned short*)(ws + 56 * MB);  // [B,H,M,64]
    unsigned short* VTh = (unsigned short*)(ws + 72 * MB);  // [B,H,64,M]
    unsigned short* AO  = (unsigned short*)(ws + 88 * MB);  // [B,N,1024]

    const int NTOKC = BATCH * SEQ_N * DIMC;   // 8M
    const int NW    = DIMC * DIMC;            // 1M
    cast_f32_bf16<<<NTOKC / 256, 256, 0, stream>>>(x,   xb,  NTOKC);
    cast_f32_bf16<<<NTOKC / 256, 256, 0, stream>>>(ctx, cb,  NTOKC);
    cast_f32_bf16<<<NW / 256,    256, 0, stream>>>(Wq,  wqb, NW);
    cast_f32_bf16<<<NW / 256,    256, 0, stream>>>(Wk,  wkb, NW);
    cast_f32_bf16<<<NW / 256,    256, 0, stream>>>(Wv,  wvb, NW);
    cast_f32_bf16<<<NW / 256,    256, 0, stream>>>(Wo,  wob, NW);

    dim3 pg(BATCH * SEQ_N / 128, DIMC / 64);  // (64, 16)
    proj_gemm<0><<<pg, 256, 0, stream>>>(xb, wqb, bq, Qh, 0.125f);  // fold D^-0.5 into Q
    proj_gemm<0><<<pg, 256, 0, stream>>>(cb, wkb, bk, Kh, 1.0f);
    proj_gemm<1><<<pg, 256, 0, stream>>>(cb, wvb, bv, VTh, 1.0f);

    dim3 ag(SEQ_N / 128, BATCH * HEADS);      // (16, 64)
    flash_attn<<<ag, 256, 0, stream>>>(Qh, Kh, VTh, AO);

    proj_gemm<2><<<pg, 256, 0, stream>>>(AO, wob, bo, out, 1.0f);
}
